// TensorFusion_67662914781607
// MI455X (gfx1250) — matrix operand order repaired and checked by reading, not compiled, not run on hardware
//
#include <hip/hip_runtime.h>

// Problem constants (from reference): B=32, N=4096, C=16, D=256
#define B_ 32
#define N_ 4096
#define C_ 16
#define D_ 256
#define NSPLIT 4
#define NCHUNK (N_ / NSPLIT)        // 1024 rows of N per workgroup
#define LDS_STRIDE (NCHUNK + 8)     // 1032 halves; 2064 B row stride (16B aligned, bank-skewed)

typedef __attribute__((ext_vector_type(16))) _Float16 v16h;
typedef __attribute__((ext_vector_type(8)))  _Float16 v8h;
typedef __attribute__((ext_vector_type(8)))  float    v8f;

// Pass 1: per (n-split, batch) workgroup. 16 waves, one 16-wide D-tile each.
// Computes partial (17 x 256) sums over its 1024-row slice using WMMA f16->f32.
__global__ __launch_bounds__(512) void fusion_pass1(
    const float* __restrict__ r, const float* __restrict__ v,
    float* __restrict__ ws) {
  __shared__ _Float16 maskLDS[17][LDS_STRIDE];  // rows 0..15: mask^T; row 16: all-below

  const int s   = blockIdx.x;       // N-split index
  const int b   = blockIdx.y;       // batch
  const int n0  = s * NCHUNK;
  const int tid = threadIdx.x;      // 0..511

  // ---- Phase 1: r read once; build transposed 0/1 mask tile in LDS ----
  for (int rr = tid; rr < NCHUNK; rr += 512) {
    const float* rp = r + ((size_t)b * N_ + (size_t)(n0 + rr)) * C_;
    bool any = false;
    #pragma unroll
    for (int c = 0; c < C_; ++c) {
      bool m = rp[c] >= 0.5f;
      any |= m;
      maskLDS[c][rr] = m ? (_Float16)1.0f : (_Float16)0.0f;
    }
    maskLDS[16][rr] = any ? (_Float16)0.0f : (_Float16)1.0f;
  }
  __syncthreads();

  const int wave = tid >> 5;        // d-tile id 0..15
  const int lane = tid & 31;
  const int col  = lane & 15;       // tile column (and A-row = category)
  const int h    = lane >> 4;       // lane half
  const int d0   = wave * 16;

  v8f   acc  = {};                  // 16x16 f32 accumulator (8 VGPRs)
  float spec = 0.0f;                // partial for the "all-below" row

  const float* vp = v + ((size_t)b * N_ + (size_t)n0) * D_ + (size_t)(d0 + col);

  #pragma unroll 2
  for (int kc = 0; kc < NCHUNK; kc += 32) {
    // --- B fragment: lane holds column d0+col, K = h*16+e (contiguous k order, same as A) ---
    // v is streamed exactly once -> non-temporal loads.
    float bf[16];
    const float* bp = vp + (size_t)(kc + h * 16) * D_;
    #pragma unroll
    for (int e = 0; e < 16; ++e) bf[e] = __builtin_nontemporal_load(bp + (size_t)e * D_);

    v16h bmat;
    #pragma unroll
    for (int e = 0; e < 16; ++e) bmat[e] = (_Float16)bf[e];

    // --- A fragment (mask^T) from LDS: row = category 'col';
    //     K = kc + h*16 + e (matches B) ---
    v8h alo = *reinterpret_cast<const v8h*>(&maskLDS[col][kc + h * 16]);
    v8h ahi = *reinterpret_cast<const v8h*>(&maskLDS[col][kc + h * 16 + 8]);
    v16h amat;
    #pragma unroll
    for (int e = 0; e < 8; ++e) { amat[e] = alo[e]; amat[8 + e] = ahi[e]; }

    // D = A x B + C  (16x16x32, f32 accumulate)
    acc = __builtin_amdgcn_wmma_f32_16x16x32_f16(
        false, amat, false, bmat, (short)0, acc, false, false);

    // --- special row: reuse the already-loaded f32 B column, exact 0/1 weights ---
    v8h s0 = *reinterpret_cast<const v8h*>(&maskLDS[16][kc + h * 16]);
    v8h s1 = *reinterpret_cast<const v8h*>(&maskLDS[16][kc + h * 16 + 8]);
    #pragma unroll
    for (int e = 0; e < 8; ++e) {
      spec = fmaf((float)s0[e], bf[e],     spec);
      spec = fmaf((float)s1[e], bf[8 + e], spec);
    }
  }

  // ---- deterministic partial writes: ws[s][b][18][256] (rows 16/17 = spec halves) ----
  float* wsp = ws + (((size_t)s * B_ + b) * 18) * D_;
  #pragma unroll
  for (int i = 0; i < 8; ++i)               // C/D layout: VGPR i -> row (i + 8*h), col
    wsp[(size_t)(i + 8 * h) * D_ + (size_t)(d0 + col)] = acc[i];
  wsp[(size_t)(16 + h) * D_ + (size_t)(d0 + col)] = spec;
}

// Pass 2: sum the NSPLIT partials, scale by 1/N, NaN-guard, overwrite all of d_out.
__global__ void fusion_pass2(const float* __restrict__ ws, float* __restrict__ out) {
  int idx = blockIdx.x * blockDim.x + threadIdx.x;
  if (idx >= B_ * 17 * D_) return;
  int d   = idx % D_;
  int row = (idx / D_) % 17;
  int b   = idx / (D_ * 17);
  float sum = 0.0f;
  #pragma unroll
  for (int s = 0; s < NSPLIT; ++s) {
    const float* p = ws + (((size_t)s * B_ + b) * 18) * D_;
    if (row < 16) sum += p[(size_t)row * D_ + d];
    else          sum += p[(size_t)16 * D_ + d] + p[(size_t)17 * D_ + d];
  }
  float val = sum * (1.0f / (float)N_);
  if (val != val) val = 0.0f;   // mirror reference's isnan -> 0
  out[idx] = val;
}

extern "C" void kernel_launch(void* const* d_in, const int* in_sizes, int n_in,
                              void* d_out, int out_size, void* d_ws, size_t ws_size,
                              hipStream_t stream) {
  const float* r = (const float*)d_in[0];   // (B,N,C) f32
  const float* v = (const float*)d_in[1];   // (B,N,D) f32
  float* out = (float*)d_out;               // (B,17,D) f32
  float* ws  = (float*)d_ws;                // needs NSPLIT*B*18*D*4 = 2,359,296 bytes

  dim3 grid1(NSPLIT, B_);
  fusion_pass1<<<grid1, 512, 0, stream>>>(r, v, ws);

  int total = B_ * 17 * D_;
  fusion_pass2<<<(total + 255) / 256, 256, 0, stream>>>(ws, out);
}
